// LocalAttentionParallel_16990890623045
// MI455X (gfx1250) — compile-verified, matrix operand order checked by reading, not run on hardware
//
#include <hip/hip_runtime.h>
#include <hip/hip_bf16.h>

// ---------------------------------------------------------------------------
// Local (sliding-window) attention, MI455X / gfx1250, wave32 + WMMA f16 + TDM.
// Pipeline:
//   1. cvt_x:  x (f32) -> Xh (f16), row-major [B*T][768]
//   2. cvt_wT: W_qkv (f32 [768][2304]) -> WhT (f16 [2304][768])  (transposed)
//   3. qkv_gemm: B-panel staged in LDS via TENSOR_LOAD_TO_LDS (double
//      buffered, s_wait_tensorcnt), WMMA f32_16x16x32_f16 accumulate.
//      Qh/Kh f16 row-major [B*T][768]; Vt f16 transposed [B][768][T].
//   4. attn: per-wave 16-row query tile; S = Q K^T (band masked, scaled),
//      staged through LDS as f16, then O = S @ V via WMMA against Vt.
//   5. ln: LayerNorm over D=768, writes final f32 output.
// ---------------------------------------------------------------------------

typedef __attribute__((ext_vector_type(16))) _Float16 v16h;
typedef __attribute__((ext_vector_type(8)))  _Float16 v8h;
typedef __attribute__((ext_vector_type(8)))  float    v8f;
typedef __attribute__((ext_vector_type(4)))  unsigned int u32x4;
typedef __attribute__((ext_vector_type(8)))  int      i32x8;
typedef __attribute__((ext_vector_type(4)))  int      i32x4;

constexpr int   BATCH  = 8;
constexpr int   TSEQ   = 2048;
constexpr int   D      = 768;
constexpr int   D3     = 2304;
constexpr int   SPAN   = 128;
constexpr float LN_EPS = 1e-5f;

constexpr size_t N_XD = (size_t)BATCH * TSEQ * D;   // 12,582,912
constexpr size_t N_W  = (size_t)D * D3;             //  1,769,472

// workspace layout (bytes)
constexpr size_t OFF_XH = 0;
constexpr size_t OFF_WT = OFF_XH + N_XD * 2;
constexpr size_t OFF_QH = OFF_WT + N_W  * 2;
constexpr size_t OFF_KH = OFF_QH + N_XD * 2;
constexpr size_t OFF_VT = OFF_KH + N_XD * 2;
constexpr size_t OFF_O  = OFF_VT + N_XD * 2;

// ---------------------------------------------------------------------------
// 16x32 f16 operand loader from row-major [row][K] storage (global or LDS).
// Per ISA 7.12.2 (16-bit A 16x32 and mirrored B 32x16 layouts):
//   lane l (l<16):  row = l,      halves 0..7 = K 0..7,  halves 8..15 = K 16..23
//   lane l (l>=16): row = l-16,   halves 0..7 = K 8..15, halves 8..15 = K 24..31
// ---------------------------------------------------------------------------
__device__ __forceinline__ v16h load_tile16x32(const _Float16* base, int ld) {
    const int lane = threadIdx.x & 31;
    const _Float16* p = base + (size_t)(lane & 15) * ld + ((lane >> 4) << 3);
    v8h lo = *(const v8h*)(p);
    v8h hi = *(const v8h*)(p + 16);
    v16h r;
#pragma unroll
    for (int i = 0; i < 8; ++i) { r[i] = lo[i]; r[i + 8] = hi[i]; }
    return r;
}

__device__ __forceinline__ v8f wmma_f16(v16h a, v16h b, v8f c) {
    return __builtin_amdgcn_wmma_f32_16x16x32_f16(
        /*neg_a=*/false, a, /*neg_b=*/false, b,
        /*c_mod=*/(short)0, c, /*reuse_a=*/false, /*reuse_b=*/false);
}

// ---------------------------------------------------------------------------
// TDM: async 2D tile load (64 rows x 64 halves) from a row-major f16 tensor
// into LDS. D# built per CDNA5 ISA ch.8 (group0 128b, group1 256b).
// This toolchain exposes the 6-arg builtin:
//   (u32x4 g0, i32x8 g1, i32x4 g2, i32x4 g3, i32x8 extra, i32 cpol)
// ---------------------------------------------------------------------------
__device__ __forceinline__ void tdm_load_tile_f16(
    unsigned int lds_addr, const _Float16* gptr,
    unsigned int tile_d0, unsigned int tile_d1,
    unsigned int tensor_d0, unsigned int tensor_d1, unsigned int stride0) {
    const unsigned long long ga = (unsigned long long)gptr;
    u32x4 g0;
    g0[0] = 1u;                                        // count=1, user mode
    g0[1] = lds_addr;                                  // LDS byte address
    g0[2] = (unsigned int)(ga & 0xFFFFFFFFu);          // global_addr[31:0]
    g0[3] = (unsigned int)((ga >> 32) & 0x01FFFFFFu)   // global_addr[56:32]
            | (2u << 30);                              // type=2 ("image")
    i32x8 g1;
    g1[0] = (int)(1u << 16);                           // data_size=1 (2 bytes)
    g1[1] = (int)((tensor_d0 & 0xFFFFu) << 16);        // tensor_dim0[15:0]
    g1[2] = (int)((tensor_d0 >> 16) | ((tensor_d1 & 0xFFFFu) << 16));
    g1[3] = (int)((tensor_d1 >> 16) | (tile_d0 << 16));// tile_dim0
    g1[4] = (int)tile_d1;                              // tile_dim1 (tile_dim2=0)
    g1[5] = (int)stride0;                              // tensor_dim0_stride lo32
    g1[6] = 0;                                         // stride0 hi / stride1 lo
    g1[7] = 0;
    i32x4 g2;                                          // dims 2/3 unused -> 1
    g2[0] = 1; g2[1] = 1; g2[2] = 0; g2[3] = 0;
    i32x4 g3;                                          // dim4 unused -> 1
    g3[0] = 0; g3[1] = (int)(1u << 16); g3[2] = 0; g3[3] = 0;
    i32x8 gx;                                          // unused trailing group
    gx[0] = 0; gx[1] = 0; gx[2] = 0; gx[3] = 0;
    gx[4] = 0; gx[5] = 0; gx[6] = 0; gx[7] = 0;
    __builtin_amdgcn_tensor_load_to_lds(g0, g1, g2, g3, gx, /*cpol=*/0);
}

// ---------------------------------------------------------------------------
// 1) x f32 -> f16
// ---------------------------------------------------------------------------
__global__ void cvt_x_kernel(const float* __restrict__ x,
                             _Float16* __restrict__ xh, size_t n) {
    size_t i = (size_t)blockIdx.x * blockDim.x + threadIdx.x;
    size_t stride = (size_t)gridDim.x * blockDim.x;
    for (; i < n; i += stride) xh[i] = (_Float16)x[i];
}

// ---------------------------------------------------------------------------
// 2) W_qkv [768][2304] f32 -> WhT [2304][768] f16 (transposed)
// ---------------------------------------------------------------------------
__global__ void cvt_wT_kernel(const float* __restrict__ w,
                              _Float16* __restrict__ wt) {
    size_t i = (size_t)blockIdx.x * blockDim.x + threadIdx.x;
    size_t stride = (size_t)gridDim.x * blockDim.x;
    for (; i < N_W; i += stride) {
        int n = (int)(i / D);
        int k = (int)(i % D);
        wt[i] = (_Float16)w[(size_t)k * D3 + n];
    }
}

// ---------------------------------------------------------------------------
// 3) QKV GEMM: [16384 x 768] @ [768 x 2304] + bias.
//    Block: 8 waves; wave tile = 16 rows x 64 cols. The shared 64x768 B-panel
//    is streamed through LDS in 64x64 chunks by the Tensor Data Mover
//    (double buffered), removing the 8x redundant per-wave B loads.
// ---------------------------------------------------------------------------
__global__ __launch_bounds__(256) void qkv_gemm_kernel(
    const _Float16* __restrict__ xh, const _Float16* __restrict__ wt,
    const float* __restrict__ bqkv,
    _Float16* __restrict__ qh, _Float16* __restrict__ kh,
    _Float16* __restrict__ vt) {
    constexpr int CH = D / 64;  // 12 K-chunks of 64
    __shared__ __align__(16) _Float16 bpanel[2][64][64];

    const int wave = threadIdx.x >> 5;
    const int lane = threadIdx.x & 31;
    const int m0 = blockIdx.y * 128 + wave * 16;   // global row (b*T + t)
    const int n0 = blockIdx.x * 64;                // global col in [0,2304)

    v8f acc[4];
#pragma unroll
    for (int t = 0; t < 4; ++t) { v8f z = {}; acc[t] = z; }

    const _Float16* abase = xh + (size_t)m0 * D;
    const _Float16* bbase = wt + (size_t)n0 * D;   // 64 rows x 768 cols panel

    // prologue: DMA chunk 0 into buffer 0 (one wave drives the TDM)
    if (wave == 0) {
        tdm_load_tile_f16((unsigned int)(unsigned long long)&bpanel[0][0][0],
                          bbase, 64, 64, (unsigned)D, 64, (unsigned)D);
    }

    for (int c = 0; c < CH; ++c) {
        if (wave == 0) {
            if (c + 1 < CH) {  // issue next chunk into the other buffer
                tdm_load_tile_f16(
                    (unsigned int)(unsigned long long)&bpanel[(c + 1) & 1][0][0],
                    bbase + (c + 1) * 64, 64, 64, (unsigned)D, 64, (unsigned)D);
                __builtin_amdgcn_s_wait_tensorcnt(1);  // chunk c landed
            } else {
                __builtin_amdgcn_s_wait_tensorcnt(0);
            }
        }
        __syncthreads();  // bpanel[c&1] visible to all waves

        const _Float16* bb = &bpanel[c & 1][0][0];
#pragma unroll
        for (int s = 0; s < 2; ++s) {
            v16h a = load_tile16x32(abase + c * 64 + s * 32, D);
            if (s == 0 && c + 1 < CH)  // warm next A slab (global_prefetch_b8)
                __builtin_prefetch(abase + (c + 1) * 64 + (size_t)(lane & 15) * D,
                                   0, 1);
#pragma unroll
            for (int t = 0; t < 4; ++t) {
                v16h b = load_tile16x32(bb + (t * 16) * 64 + s * 32, 64);
                acc[t] = wmma_f16(a, b, acc[t]);
            }
        }
        __syncthreads();  // all waves done with bpanel[c&1]; TDM may overwrite
    }

    // C layout: lane l holds N = l%16, rows M = 8*(l/16) + r for VGPR r.
    const int mrow = 8 * (lane >> 4);
#pragma unroll
    for (int t = 0; t < 4; ++t) {
        const int n = n0 + t * 16 + (lane & 15);
        const float bias = bqkv[n];
        if (n < D) {                        // Q  (wave-uniform branch)
            _Float16* dst = qh + n;
#pragma unroll
            for (int r = 0; r < 8; ++r)
                dst[(size_t)(m0 + mrow + r) * D] = (_Float16)(acc[t][r] + bias);
        } else if (n < 2 * D) {             // K
            _Float16* dst = kh + (n - D);
#pragma unroll
            for (int r = 0; r < 8; ++r)
                dst[(size_t)(m0 + mrow + r) * D] = (_Float16)(acc[t][r] + bias);
        } else {                            // V, transposed [b][d][t]
            const int bidx = m0 / TSEQ;
            const int trow = m0 % TSEQ;
            _Float16* dst =
                vt + ((size_t)bidx * D + (n - 2 * D)) * TSEQ + trow + mrow;
#pragma unroll
            for (int r = 0; r < 8; ++r)
                dst[r] = (_Float16)(acc[t][r] + bias);
        }
    }
}

// ---------------------------------------------------------------------------
// 4) Banded attention. One wave per 16-row query tile.
//    S tile: 9 j-tiles (144 band cols), masked+scaled, staged in LDS (f16,
//    zero-padded to 160 cols). Then O = S @ V over 48 n-tiles x 5 K-blocks.
// ---------------------------------------------------------------------------
__global__ __launch_bounds__(256) void attn_kernel(
    const _Float16* __restrict__ qh, const _Float16* __restrict__ kh,
    const _Float16* __restrict__ vt, float* __restrict__ o) {
    constexpr int SW = 160;  // padded S width (143 valid band cols -> 160)
    __shared__ __align__(16) _Float16 s_s[8][16][SW];

    const int wave  = threadIdx.x >> 5;
    const int lane  = threadIdx.x & 31;
    const int gtile = blockIdx.x * 8 + wave;   // 0..1023
    const int grow0 = gtile * 16;              // global row (b*T + t)
    const int b     = grow0 / TSEQ;
    const int i0    = grow0 % TSEQ;

    const float scale = rsqrtf((float)(D * SPAN));
    const int mrow = 8 * (lane >> 4);
    const int ncol = lane & 15;

    // zero this wave's S strip (10 x 16B per lane)
    {
        v8h z = {};
        _Float16* sp = &s_s[wave][0][0];
        for (int idx = lane * 8; idx < 16 * SW; idx += 32 * 8)
            *(v8h*)(sp + idx) = z;
    }

    // --- phase 1: S = (Q K^T) * scale, band-masked ---
    const int jt_start = (i0 >= SPAN) ? 0 : (8 - (i0 >> 4));
    const _Float16* qa = qh + (size_t)grow0 * D;
    for (int jt = jt_start; jt < 9; ++jt) {
        const int j0 = i0 - SPAN + jt * 16;    // >= 0 for jt >= jt_start
        const _Float16* kb = kh + ((size_t)b * TSEQ + j0) * D;
        v8f s = {};
        for (int kk = 0; kk < D; kk += 32) {
            v16h a  = load_tile16x32(qa + kk, D);
            v16h bm = load_tile16x32(kb + kk, D);
            s = wmma_f16(a, bm, s);
        }
        const int jcol  = jt * 16 + ncol;
        const int jglob = j0 + ncol;
#pragma unroll
        for (int r = 0; r < 8; ++r) {
            const int i   = i0 + mrow + r;
            const int dlt = i - jglob;
            const float val = (dlt >= 0 && dlt < SPAN) ? s[r] * scale : 0.0f;
            s_s[wave][mrow + r][jcol] = (_Float16)val;
        }
    }
    __syncthreads();

    // --- phase 2: O = S @ V (Vt is [b][d][t]) ---
    const int jbase = i0 - SPAN;
    const int koff  = (lane >> 4) << 3;  // 0 or 8
#pragma unroll 4
    for (int d0 = 0; d0 < D; d0 += 16) {
        v8f oacc = {};
#pragma unroll
        for (int kb2 = 0; kb2 < 5; ++kb2) {
            // A from LDS (S strip, zero-padded)
            const _Float16* ap = &s_s[wave][lane & 15][kb2 * 32 + koff];
            v8h alo = *(const v8h*)(ap);
            v8h ahi = *(const v8h*)(ap + 16);
            v16h a;
#pragma unroll
            for (int ii = 0; ii < 8; ++ii) { a[ii] = alo[ii]; a[ii + 8] = ahi[ii]; }
            // B from Vt; clamp columns into [0, T-8] (S is zero there anyway)
            int c0 = jbase + kb2 * 32 + koff;
            int c1 = c0 + 16;
            c0 = c0 < 0 ? 0 : (c0 > TSEQ - 8 ? TSEQ - 8 : c0);
            c1 = c1 < 0 ? 0 : (c1 > TSEQ - 8 ? TSEQ - 8 : c1);
            const _Float16* vb =
                vt + ((size_t)b * D + d0 + (lane & 15)) * TSEQ;
            v8h blo = *(const v8h*)(vb + c0);
            v8h bhi = *(const v8h*)(vb + c1);
            v16h bm;
#pragma unroll
            for (int ii = 0; ii < 8; ++ii) { bm[ii] = blo[ii]; bm[ii + 8] = bhi[ii]; }
            oacc = wmma_f16(a, bm, oacc);
        }
#pragma unroll
        for (int r = 0; r < 8; ++r)
            o[((size_t)grow0 + mrow + r) * D + d0 + ncol] = oacc[r];
    }
}

// ---------------------------------------------------------------------------
// 5) LayerNorm over D=768, one block per row.
// ---------------------------------------------------------------------------
__global__ __launch_bounds__(256) void ln_kernel(
    const float* __restrict__ o, const float* __restrict__ lw,
    const float* __restrict__ lb, float* __restrict__ out) {
    __shared__ float red[256];
    const int row = blockIdx.x;
    const int tid = threadIdx.x;
    const float* p = o + (size_t)row * D;
    const float v0 = p[tid], v1 = p[tid + 256], v2 = p[tid + 512];

    red[tid] = v0 + v1 + v2;
    __syncthreads();
    for (int off = 128; off > 0; off >>= 1) {
        if (tid < off) red[tid] += red[tid + off];
        __syncthreads();
    }
    const float mu = red[0] * (1.0f / D);
    __syncthreads();

    const float d0 = v0 - mu, d1 = v1 - mu, d2 = v2 - mu;
    red[tid] = d0 * d0 + d1 * d1 + d2 * d2;
    __syncthreads();
    for (int off = 128; off > 0; off >>= 1) {
        if (tid < off) red[tid] += red[tid + off];
        __syncthreads();
    }
    const float inv = rsqrtf(red[0] * (1.0f / D) + LN_EPS);

    float* q = out + (size_t)row * D;
    q[tid]       = d0 * inv * lw[tid]       + lb[tid];
    q[tid + 256] = d1 * inv * lw[tid + 256] + lb[tid + 256];
    q[tid + 512] = d2 * inv * lw[tid + 512] + lb[tid + 512];
}

// ---------------------------------------------------------------------------
extern "C" void kernel_launch(void* const* d_in, const int* in_sizes, int n_in,
                              void* d_out, int out_size, void* d_ws,
                              size_t ws_size, hipStream_t stream) {
    (void)in_sizes; (void)n_in; (void)out_size; (void)ws_size;
    const float* x    = (const float*)d_in[0];
    const float* Wqkv = (const float*)d_in[1];
    const float* bqkv = (const float*)d_in[2];
    const float* lnw  = (const float*)d_in[3];
    const float* lnb  = (const float*)d_in[4];

    char* ws = (char*)d_ws;
    _Float16* xh = (_Float16*)(ws + OFF_XH);
    _Float16* wt = (_Float16*)(ws + OFF_WT);
    _Float16* qh = (_Float16*)(ws + OFF_QH);
    _Float16* kh = (_Float16*)(ws + OFF_KH);
    _Float16* vt = (_Float16*)(ws + OFF_VT);
    float*    ob = (float*)   (ws + OFF_O);

    cvt_x_kernel <<<1024, 256, 0, stream>>>(x, xh, N_XD);
    cvt_wT_kernel<<<1024, 256, 0, stream>>>(Wqkv, wt);

    dim3 g1(D3 / 64, (BATCH * TSEQ) / 128);
    qkv_gemm_kernel<<<g1, 256, 0, stream>>>(xh, wt, bqkv, qh, kh, vt);

    attn_kernel<<<(BATCH * TSEQ / 16) / 8, 256, 0, stream>>>(qh, kh, vt, ob);

    ln_kernel<<<BATCH * TSEQ, 256, 0, stream>>>(ob, lnw, lnb, (float*)d_out);
}
